// PointCloudEncoder_70806830842024
// MI455X (gfx1250) — compile-verified, matrix operand order checked by reading, not run on hardware
//
#include <hip/hip_runtime.h>
#include <hip/hip_bf16.h>
#include <math.h>
#include <stdint.h>

// ---------------------------------------------------------------------------
// PointCloudEncoder for MI455X (gfx1250, wave32, WMMA).
// GEMMs: bf16 WMMA (v_wmma_f32_16x16x32_bf16), f32 accumulate,
// 128x128 block tiles, double-buffered LDS, async global->LDS staging.
// ---------------------------------------------------------------------------

typedef __bf16 bf16;
typedef __attribute__((ext_vector_type(16))) __bf16 v16bf;
typedef __attribute__((ext_vector_type(8)))  __bf16 v8bf;
typedef __attribute__((ext_vector_type(8)))  float  v8f;

#define EPSV    1e-8f
#define PTS_M   32768        // B*N = 16*2048
#define OUT_W   3587

#if defined(__has_builtin)
#  if __has_builtin(__builtin_amdgcn_global_load_async_to_lds_b128)
#    define HAVE_ASYNC_LDS 1
#  endif
#  if __has_builtin(__builtin_amdgcn_s_wait_asynccnt)
#    define HAVE_WAIT_ASYNC 1
#  endif
#endif

// Builtin parameter types (from hipcc diagnostic):
//   arg0: GCC vector 'int __attribute__((vector_size(16)))' in addrspace(1) (global src)
//   arg1: same pointee in addrspace(3) (LDS dst)
typedef int v4i_g __attribute__((vector_size(16)));
typedef __attribute__((address_space(1))) v4i_g gv4i;
typedef __attribute__((address_space(3))) v4i_g lv4i;

// 16B global -> LDS copy; async (ASYNCcnt) when available, sync fallback.
__device__ inline void cp16(const void* g, void* lds) {
#ifdef HAVE_ASYNC_LDS
  __builtin_amdgcn_global_load_async_to_lds_b128(
      (gv4i*)(uintptr_t)g, (lv4i*)(uintptr_t)lds, 0, 0);
#else
  *(int4*)lds = *(const int4*)g;
#endif
}

__device__ inline void wait_async_lds() {
#ifdef HAVE_ASYNC_LDS
#  ifdef HAVE_WAIT_ASYNC
  __builtin_amdgcn_s_wait_asynccnt(0);
#  else
  asm volatile("s_wait_asynccnt 0x0" ::: "memory");
#  endif
#endif
}

// ---------------------------------------------------------------------------
// helpers
// ---------------------------------------------------------------------------
__device__ inline float wave_sum32(float v) {
#pragma unroll
  for (int o = 16; o; o >>= 1) v += __shfl_xor(v, o, 32);
  return v;
}

// Load a 16-element bf16 WMMA fragment from LDS.
// ISA layout (16-bit A 16x32 / B mirrored): lane half selects K-group;
// per lane: K {half*8 .. half*8+7} then K {16+half*8 .. 16+half*8+7}.
__device__ inline v16bf load_frag16(const bf16* base) {
  v8bf lo = *(const v8bf*)(base);
  v8bf hi = *(const v8bf*)(base + 16);
  v16bf r;
#pragma unroll
  for (int i = 0; i < 8; ++i) { r[i] = lo[i]; r[i + 8] = hi[i]; }
  return r;
}

// ---------------------------------------------------------------------------
// Weight convert + transpose: W (K x N) f32 -> Wt (N x K) bf16
// ---------------------------------------------------------------------------
__global__ void k_conv_transpose(const float* __restrict__ W, bf16* __restrict__ Wt,
                                 int K, int N) {
  int idx = blockIdx.x * 256 + threadIdx.x;
  if (idx >= K * N) return;
  int k = idx / N, n = idx - k * N;
  Wt[(size_t)n * K + k] = (bf16)W[(size_t)k * N + n];
}

// ---------------------------------------------------------------------------
// Stage-0: h = relu(pc @ w1_0 + b1_0), K=3, N=256 -> bf16
// ---------------------------------------------------------------------------
__global__ __launch_bounds__(256) void k_stage0(const float* __restrict__ pc,
                                                const float* __restrict__ w1,
                                                const float* __restrict__ b1,
                                                bf16* __restrict__ out) {
  int m = blockIdx.x;
  int c = threadIdx.x;
  float x0 = pc[m * 3 + 0], x1 = pc[m * 3 + 1], x2 = pc[m * 3 + 2];
  float v = fmaf(x0, w1[c], fmaf(x1, w1[256 + c], fmaf(x2, w1[512 + c], b1[c])));
  out[(size_t)m * 256 + c] = (bf16)fmaxf(v, 0.f);
}

// ---------------------------------------------------------------------------
// WMMA GEMM:  out = [relu](A(MxK,bf16) @ Bt(NxK,bf16)^T + bias)
// block tile 128(M) x 128(N); 8 waves (4x2), wave tile 32x64 -> 8 WMMA/k-step
// double-buffered LDS (2 x (8KB A + 8KB B)), async global->LDS staging
// ---------------------------------------------------------------------------
__global__ __launch_bounds__(256)
void k_wmma_gemm(const bf16* __restrict__ A, const bf16* __restrict__ Bt,
                 const float* __restrict__ bias,
                 float* outF, int ldF, int colOff,
                 bf16* outB, int N, int K, int doRelu) {
  __shared__ __align__(16) bf16 As[2][128 * 32];   // 2 x 8 KB
  __shared__ __align__(16) bf16 Bs[2][128 * 32];   // 2 x 8 KB

  const int tid  = threadIdx.x;
  const int wave = tid >> 5, lane = tid & 31;
  const int half = lane >> 4, lr = lane & 15;
  const int wm   = wave >> 1, wn = wave & 1;          // 4x2 wave grid
  const long m0  = (long)blockIdx.y * 128;
  const long n0  = (long)blockIdx.x * 128;

  const int cr  = tid >> 1;                           // copy row 0..127
  const int seg = (tid & 1) * 32;                     // 32B segment in 64B row

  v8f z = {0.f, 0.f, 0.f, 0.f, 0.f, 0.f, 0.f, 0.f};
  v8f acc[2][4] = {{z, z, z, z}, {z, z, z, z}};

  const int T = K >> 5;                               // k-tiles of 32

  auto stage = [&](int t, int buf) {
    const char* ga = (const char*)A  + (((m0 + cr) * (long)K + ((long)t << 5)) << 1) + seg;
    const char* gb = (const char*)Bt + (((n0 + cr) * (long)K + ((long)t << 5)) << 1) + seg;
    char* la = (char*)&As[buf][0] + cr * 64 + seg;
    char* lb = (char*)&Bs[buf][0] + cr * 64 + seg;
    cp16(ga,      la);
    cp16(ga + 16, la + 16);
    cp16(gb,      lb);
    cp16(gb + 16, lb + 16);
  };

  stage(0, 0);
  wait_async_lds();
  __syncthreads();

  for (int t = 0; t < T; ++t) {
    const int cur = t & 1;
    if (t + 1 < T) stage(t + 1, cur ^ 1);             // overlap next tile load

    const bf16* Ab = &As[cur][0];
    const bf16* Bb = &Bs[cur][0];
    v16bf a0 = load_frag16(&Ab[(32 * wm +      lr) * 32 + half * 8]);
    v16bf a1 = load_frag16(&Ab[(32 * wm + 16 + lr) * 32 + half * 8]);
    v16bf b[4];
#pragma unroll
    for (int j = 0; j < 4; ++j)
      b[j] = load_frag16(&Bb[(64 * wn + 16 * j + lr) * 32 + half * 8]);

#pragma unroll
    for (int j = 0; j < 4; ++j) {
      acc[0][j] = __builtin_amdgcn_wmma_f32_16x16x32_bf16(false, a0, false, b[j], (short)0, acc[0][j], false, false);
      acc[1][j] = __builtin_amdgcn_wmma_f32_16x16x32_bf16(false, a1, false, b[j], (short)0, acc[1][j], false, false);
    }

    if (t + 1 < T) {
      wait_async_lds();
      __syncthreads();
    }
  }

  // epilogue: C layout -> vgpr r = row (half*8 + r), lane lr = col
#pragma unroll
  for (int i = 0; i < 2; ++i) {
#pragma unroll
    for (int j = 0; j < 4; ++j) {
      long mb = m0 + 32 * wm + 16 * i + half * 8;
      long n  = n0 + 64 * wn + 16 * j + lr;
      float bv = bias[n];
#pragma unroll
      for (int r = 0; r < 8; ++r) {
        float v = acc[i][j][r] + bv;
        if (doRelu) v = fmaxf(v, 0.f);
        long m = mb + r;
        if (outF) outF[m * (long)ldF + colOff + n] = v;
        if (outB) outB[m * (long)N + n] = (bf16)v;
      }
    }
  }
}

// ---------------------------------------------------------------------------
// Per-(b,c) image stats: mean and sqrt(var_ddof1 + eps); one wave per (b,c)
// ---------------------------------------------------------------------------
__global__ __launch_bounds__(32) void k_img_stats(const float* __restrict__ img,
                                                  float* __restrict__ mu,
                                                  float* __restrict__ sc,
                                                  int S) {
  int bc = blockIdx.x;
  int n  = S * S;
  const float* p = img + (size_t)bc * n;
  float s = 0.f, ss = 0.f;
  for (int i = threadIdx.x; i < n; i += 32) {
    float v = p[i];
    s += v; ss += v * v;
  }
  s  = wave_sum32(s);
  ss = wave_sum32(ss);
  if (threadIdx.x == 0) {
    float m   = s / (float)n;
    float var = (ss - (float)n * m * m) / (float)(n - 1);
    mu[bc] = m;
    sc[bc] = sqrtf(var + EPSV);
  }
}

// ---------------------------------------------------------------------------
// AdaIN normalize: per point m (one wave): x_norm = (x-mu)/sqrt(var+eps),
// out_bf16 = (x_norm + im_mu[b,c]) * im_scale[b,c]
// ---------------------------------------------------------------------------
__global__ __launch_bounds__(256) void k_adain_norm(const float* __restrict__ enc,
                                                    const float* __restrict__ imMu,
                                                    const float* __restrict__ imSc,
                                                    bf16* __restrict__ outN, int F) {
  int wave = threadIdx.x >> 5, lane = threadIdx.x & 31;
  int m = blockIdx.x * 8 + wave;
  int b = m >> 11;                       // m / 2048
  const float* row = enc + (size_t)m * F;
  float s = 0.f, ss = 0.f;
  for (int c = lane; c < F; c += 32) {
    float v = row[c];
    s += v; ss += v * v;
  }
  s  = wave_sum32(s);
  ss = wave_sum32(ss);
  float mean = s / (float)F;
  float var  = (ss - (float)F * mean * mean) / (float)(F - 1);
  float inv  = 1.f / sqrtf(var + EPSV);
  const float* mu = imMu + (size_t)b * F;
  const float* sc = imSc + (size_t)b * F;
  for (int c = lane; c < F; c += 32) {
    float v = (row[c] - mean) * inv + mu[c];
    outN[(size_t)m * F + c] = (bf16)(v * sc[c]);
  }
}

// ---------------------------------------------------------------------------
// Bilinear projection gather; one wave per point, lanes stride channels
// ---------------------------------------------------------------------------
__global__ __launch_bounds__(256) void k_project(const float* __restrict__ img,
                                                 const float* __restrict__ pc,
                                                 float* __restrict__ out,
                                                 int C, int S, int colOff) {
  int wave = threadIdx.x >> 5, lane = threadIdx.x & 31;
  int m = blockIdx.x * 8 + wave;
  int b = m >> 11;
  float X = pc[m * 3 + 0], Y = pc[m * 3 + 1], Z = pc[m * 3 + 2];
  float h = fminf(fmaxf(248.f * Y / Z + 111.5f, 0.f), 223.f);
  float w = fminf(fmaxf(248.f * (-X) / Z + 111.5f, 0.f), 223.f);
  float x = h * (float)(S - 1) / 223.f;
  float y = w * (float)(S - 1) / 223.f;
  float x1 = floorf(x), x2 = ceilf(x);
  float y1 = floorf(y), y2 = ceilf(y);
  int xi1 = (int)x1, xi2 = (int)x2, yi1 = (int)y1, yi2 = (int)y2;
  float w11 = (x2 - x) * (y2 - y);
  float w21 = (x - x1) * (y2 - y);
  float w12 = (x2 - x) * (y - y1);
  float w22 = (x - x1) * (y - y1);
  const float* base = img + (size_t)b * C * S * S;
  float* orow = out + (size_t)m * OUT_W + colOff;
  for (int c = lane; c < C; c += 32) {
    const float* pch = base + (size_t)c * S * S;
    float q11 = pch[xi1 * S + yi1];
    float q12 = pch[xi1 * S + yi2];
    float q21 = pch[xi2 * S + yi1];
    float q22 = pch[xi2 * S + yi2];
    orow[c] = q11 * w11 + q12 * w12 + q21 * w21 + q22 * w22;
  }
}

// ---------------------------------------------------------------------------
// copy init_pc into last 3 output columns
// ---------------------------------------------------------------------------
__global__ void k_copy_pc(const float* __restrict__ pc, float* __restrict__ out) {
  int i = blockIdx.x * 256 + threadIdx.x;
  if (i >= PTS_M * 3) return;
  int m = i / 3, d = i - m * 3;
  out[(size_t)m * OUT_W + 3584 + d] = pc[i];
}

// ---------------------------------------------------------------------------
// host launcher
// ---------------------------------------------------------------------------
extern "C" void kernel_launch(void* const* d_in, const int* in_sizes, int n_in,
                              void* d_out, int out_size, void* d_ws, size_t ws_size,
                              hipStream_t stream) {
  (void)in_sizes; (void)n_in; (void)out_size; (void)ws_size;
  const float* pc     = (const float*)d_in[0];
  const float* img[3] = {(const float*)d_in[1], (const float*)d_in[2], (const float*)d_in[3]};
  const float* w1[3]  = {(const float*)d_in[4],  (const float*)d_in[10], (const float*)d_in[16]};
  const float* b1[3]  = {(const float*)d_in[5],  (const float*)d_in[11], (const float*)d_in[17]};
  const float* w2[3]  = {(const float*)d_in[6],  (const float*)d_in[12], (const float*)d_in[18]};
  const float* b2[3]  = {(const float*)d_in[7],  (const float*)d_in[13], (const float*)d_in[19]};
  const float* wf[3]  = {(const float*)d_in[8],  (const float*)d_in[14], (const float*)d_in[20]};
  const float* bf_[3] = {(const float*)d_in[9],  (const float*)d_in[15], (const float*)d_in[21]};
  float* out = (float*)d_out;

  const int F[3] = {256, 512, 1024};
  const int S[3] = {56, 28, 14};
  const int M = PTS_M;

  // ---- carve scratch ----
  char* ws = (char*)d_ws;
  size_t cur = 0;
  auto carve = [&](size_t bytes) -> char* {
    char* p = ws + cur;
    cur = (cur + bytes + 255) & ~(size_t)255;
    return p;
  };
  bf16* wt_w2[3], *wt_wf[3];
  for (int i = 0; i < 3; ++i) wt_w2[i] = (bf16*)carve((size_t)F[i] * F[i] * 2);
  bf16* wt_w1b[2];
  wt_w1b[0] = (bf16*)carve((size_t)512 * 256 * 2);    // w1_1^T
  wt_w1b[1] = (bf16*)carve((size_t)1024 * 512 * 2);   // w1_2^T
  for (int i = 0; i < 3; ++i) wt_wf[i] = (bf16*)carve((size_t)F[i] * F[i] * 2);
  bf16* h1buf = (bf16*)carve((size_t)M * 1024 * 2);   // reused activations
  float* encF[3]; bf16* encB[3];
  for (int i = 0; i < 3; ++i) {
    encF[i] = (float*)carve((size_t)M * F[i] * 4);
    encB[i] = (bf16*)carve((size_t)M * F[i] * 2);
  }
  float* imMu[3]; float* imSc[3];
  for (int i = 0; i < 3; ++i) {
    imMu[i] = (float*)carve((size_t)16 * F[i] * 4);
    imSc[i] = (float*)carve((size_t)16 * F[i] * 4);
  }

  // ---- weight conversion (f32 -> bf16, transposed) ----
  auto convT = [&](const float* W, bf16* Wt, int K, int N) {
    int n = K * N;
    k_conv_transpose<<<(n + 255) / 256, 256, 0, stream>>>(W, Wt, K, N);
  };
  for (int i = 0; i < 3; ++i) convT(w2[i], wt_w2[i], F[i], F[i]);
  convT(w1[1], wt_w1b[0], 256, 512);
  convT(w1[2], wt_w1b[1], 512, 1024);
  for (int i = 0; i < 3; ++i) convT(wf[i], wt_wf[i], F[i], F[i]);

  // ---- image statistics ----
  for (int i = 0; i < 3; ++i)
    k_img_stats<<<16 * F[i], 32, 0, stream>>>(img[i], imMu[i], imSc[i], S[i]);

  // ---- encoder ----
  k_stage0<<<M, 256, 0, stream>>>(pc, w1[0], b1[0], h1buf);

  auto gemm = [&](const bf16* A, const bf16* Bt, const float* bias,
                  float* oF, int ldF, int colOff, bf16* oB,
                  int N, int K, int relu) {
    dim3 grid(N / 128, M / 128);
    k_wmma_gemm<<<grid, 256, 0, stream>>>(A, Bt, bias, oF, ldF, colOff, oB, N, K, relu);
  };

  gemm(h1buf,   wt_w2[0],  b2[0], encF[0], 256,  0, encB[0], 256,  256,  1); // encoded0
  gemm(encB[0], wt_w1b[0], b1[1], nullptr, 0,    0, h1buf,   512,  256,  1); // h1_1
  gemm(h1buf,   wt_w2[1],  b2[1], encF[1], 512,  0, encB[1], 512,  512,  1); // encoded1
  gemm(encB[1], wt_w1b[1], b1[2], nullptr, 0,    0, h1buf,   1024, 512,  1); // h1_2
  gemm(h1buf,   wt_w2[2],  b2[2], encF[2], 1024, 0, encB[2], 1024, 1024, 1); // encoded2

  // ---- AdaIN: normalize then wf GEMM straight into concat output ----
  const int colA[3] = {0, 256, 768};
  for (int i = 0; i < 3; ++i) {
    k_adain_norm<<<M / 8, 256, 0, stream>>>(encF[i], imMu[i], imSc[i], h1buf, F[i]);
    gemm(h1buf, wt_wf[i], bf_[i], out, OUT_W, colA[i], nullptr, F[i], F[i], 0);
  }

  // ---- bilinear projection into concat output ----
  const int colP[3] = {1792, 2048, 2560};
  for (int i = 0; i < 3; ++i)
    k_project<<<M / 8, 256, 0, stream>>>(img[i], pc, out, F[i], S[i], colP[i]);

  // ---- init_pc passthrough ----
  k_copy_pc<<<(PTS_M * 3 + 255) / 256, 256, 0, stream>>>(pc, out);
}
